// HeteroGAT_64613488001780
// MI455X (gfx1250) — matrix operand order repaired and checked by reading, not compiled, not run on hardware
//
#include <hip/hip_runtime.h>
#include <stdint.h>

// ---------------- types / helpers ----------------
typedef __attribute__((ext_vector_type(16))) __bf16 v16bf;
typedef __attribute__((ext_vector_type(8)))  float  v8f;

union Frag16 { v16bf v; unsigned short s[16]; uint4 q[2]; };

__device__ __forceinline__ unsigned short f2bf(float f) {
    unsigned u = __float_as_uint(f);
    u += 0x7FFFu + ((u >> 16) & 1u);          // round-to-nearest-even
    return (unsigned short)(u >> 16);
}
// monotone float<->uint encoding so atomicMax(uint) == float max; key 0 == -inf
__device__ __forceinline__ unsigned fenc(float f) {
    unsigned u = __float_as_uint(f);
    return (u & 0x80000000u) ? ~u : (u | 0x80000000u);
}
__device__ __forceinline__ float fdec(unsigned u) {
    return __uint_as_float((u & 0x80000000u) ? (u ^ 0x80000000u) : ~u);
}

// ---------------- bulk f32 -> bf16 convert (one-time, off the GEMM hot path) ----------------
__global__ __launch_bounds__(256) void k_f2bf(const float* __restrict__ X,
                                              unsigned short* __restrict__ Y, long n) {
    long i = ((long)blockIdx.x * 256 + threadIdx.x) * 4;
    if (i >= n) return;
    float4 f = *reinterpret_cast<const float4*>(X + i);
    ushort4 y;
    y.x = f2bf(f.x); y.y = f2bf(f.y); y.z = f2bf(f.z); y.w = f2bf(f.w);
    *reinterpret_cast<ushort4*>(Y + i) = y;
}

// ---------------- weight prep ----------------
// W [3,K,N] f32  ->  Wt [3,N,K] bf16 (transposed so B-fragments load contiguously)
__global__ __launch_bounds__(256) void k_wt_bf16(const float* __restrict__ W,
                                                 unsigned short* __restrict__ Wt,
                                                 int K, int N) {
    long i = (long)blockIdx.x * 256 + threadIdx.x;
    long total = 3L * K * N;
    if (i >= total) return;
    int r = (int)(i / (K * N));
    int rem = (int)(i % (K * N));
    int n = rem / K, k = rem % K;
    Wt[i] = f2bf(W[((long)r * K + k) * N + n]);
}

// wtilde[r][h][k] = sum_c W[r][k][h*C+c] * a[r][h][c]   (folds W@a so hd is never built)
__global__ __launch_bounds__(256) void k_wtilde(const float* __restrict__ W,
                                                const float* __restrict__ a,
                                                float* __restrict__ wt,
                                                int K, int HC, int C) {
    int i = blockIdx.x * 256 + threadIdx.x;
    int total = 3 * 4 * K;
    if (i >= total) return;
    int r = i / (4 * K);
    int h = (i / K) & 3;
    int k = i % K;
    const float* w  = W + ((long)r * K + k) * HC + h * C;
    const float* av = a + ((long)r * 4 + h) * C;
    float s = 0.f;
    for (int c = 0; c < C; ++c) s += w[c] * av[c];
    wt[i] = s;
}

// ---------------- WMMA GEMM: Hout[M,NCOLS] = X[M,128] @ W[128,NCOLS] ----------------
// X bf16 row-major [M,128], Wt bf16 [NCOLS,128] (pre-transposed). Pure b128 loads + WMMA.
template <int NCOLS>
__global__ __launch_bounds__(256) void k_gemm(const unsigned short* __restrict__ X,
                                              const unsigned short* __restrict__ Wt,
                                              float* __restrict__ Hout, int M) {
    constexpr int NT = NCOLS / 16;
    const int wave = threadIdx.x >> 5;
    const int lane = threadIdx.x & 31;
    const int m0 = blockIdx.x * 128 + wave * 16;
    if (m0 >= M) return;                        // wave-uniform exit (EXEC all-ones in WMMA)

    const int rsel = lane & 15;
    const int kbA  = (lane >> 4) << 3;          // A: K 0-7/16-23 vs 8-15/24-31 per lane half
    const int kbB  = (lane >> 4) << 3;          // B: K 0-7/16-23 vs 8-15/24-31 per lane half
    int row = m0 + rsel; if (row >= M) row = M - 1;
    const unsigned short* xrow = X + (long)row * 128;

    v8f acc[NT] = {};

#pragma unroll
    for (int k0 = 0; k0 < 128; k0 += 32) {
        Frag16 a;
        a.q[0] = *reinterpret_cast<const uint4*>(xrow + k0 + kbA);
        a.q[1] = *reinterpret_cast<const uint4*>(xrow + k0 + kbA + 16);
#pragma unroll
        for (int j = 0; j < NT; ++j) {
            Frag16 b;
            const unsigned short* wrow = Wt + (long)(j * 16 + rsel) * 128 + k0 + kbB;
            b.q[0] = *reinterpret_cast<const uint4*>(wrow);
            b.q[1] = *reinterpret_cast<const uint4*>(wrow + 16);
            acc[j] = __builtin_amdgcn_wmma_f32_16x16x32_bf16(
                false, a.v, false, b.v, (short)0, acc[j], false, false);
        }
    }
    const int radd = (lane >> 4) * 8;           // D layout: VGPR r -> rows r / r+8
#pragma unroll
    for (int j = 0; j < NT; ++j) {
#pragma unroll
        for (int r = 0; r < 8; ++r) {
            int orow = m0 + r + radd;
            if (orow < M) Hout[(long)orow * NCOLS + j * 16 + rsel] = acc[j][r];
        }
    }
}

// ---------------- attention scalar: e[n,h] = X[n,:] . wt[h,:]  (K=128) ----------------
__global__ __launch_bounds__(256) void k_escalar(const float* __restrict__ X,
                                                 const float* __restrict__ wt,
                                                 float* __restrict__ e, int N) {
    int i = blockIdx.x * 256 + threadIdx.x;
    if (i >= N * 4) return;
    int n = i >> 2, h = i & 3;
    const float* x = X + (long)n * 128;
    const float* w = wt + h * 128;
    float s = 0.f;
#pragma unroll 8
    for (int k = 0; k < 128; ++k) s += x[k] * w[k];
    e[i] = s;
}

// ---------------- edge pass 1: segment-max via ordered-uint atomicMax ----------------
__global__ __launch_bounds__(256) void k_edge_max(const int* __restrict__ src,
                                                  const int* __restrict__ dst,
                                                  const float* __restrict__ es,
                                                  const float* __restrict__ ed,
                                                  unsigned* __restrict__ mkey, int E) {
    int e = blockIdx.x * 256 + threadIdx.x;
    if (e >= E) return;
    int s = src[e], d = dst[e];
#pragma unroll
    for (int h = 0; h < 4; ++h) {
        float v = es[s * 4 + h] + ed[d * 4 + h];
        v = v > 0.f ? v : 0.2f * v;             // leaky_relu(0.2)
        atomicMax(&mkey[(long)d * 4 + h], fenc(v));
    }
}

// ---------------- edge pass 2: num += exp(e-m)*hs[src], den += exp(e-m) ----------------
// one wave32 per edge; lane owns CH/32 channels
template <int CH>
__global__ __launch_bounds__(256) void k_edge_scatter(const int* __restrict__ src,
                                                      const int* __restrict__ dst,
                                                      const float* __restrict__ es,
                                                      const float* __restrict__ ed,
                                                      const unsigned* __restrict__ mkey,
                                                      const float* __restrict__ hs,
                                                      float* __restrict__ num,
                                                      float* __restrict__ den, int E) {
    int e = blockIdx.x * 8 + (threadIdx.x >> 5);
    if (e >= E) return;                          // wave-uniform
    int lane = threadIdx.x & 31;
    int s = src[e], d = dst[e];
    __builtin_prefetch(hs + (long)s * CH + lane, 0, 0);
    float ee[4];
#pragma unroll
    for (int h = 0; h < 4; ++h) {
        float v = es[s * 4 + h] + ed[d * 4 + h];
        v = v > 0.f ? v : 0.2f * v;
        float m = fdec(mkey[(long)d * 4 + h]);
        if (!__builtin_isfinite(m)) m = 0.f;     // matches where(isfinite(m), m, 0)
        ee[h] = __expf(v - m);
    }
    if (lane < 4) atomicAdd(&den[(long)d * 4 + lane], ee[lane]);
    constexpr int CPH = CH / 4;
#pragma unroll
    for (int j = 0; j < CH / 32; ++j) {
        int c = j * 32 + lane;
        int h = c / CPH;
        atomicAdd(&num[(long)d * CH + c], ee[h] * hs[(long)s * CH + c]);
    }
}

// ---------------- finalize: out = numA/denA + bA (+ numB/denB + bB) [+relu] ----------------
__global__ __launch_bounds__(256) void k_finalize(const float* __restrict__ numA,
                                                  const float* __restrict__ denA,
                                                  const float* __restrict__ bA,
                                                  const float* __restrict__ numB,
                                                  const float* __restrict__ denB,
                                                  const float* __restrict__ bB,
                                                  float* __restrict__ out,
                                                  int N, int CH, int relu) {
    long i = (long)blockIdx.x * 256 + threadIdx.x;
    if (i >= (long)N * CH) return;
    int n = (int)(i / CH), c = (int)(i % CH);
    int h = c / (CH / 4);
    float v = numA[i] / (denA[(long)n * 4 + h] + 1e-16f) + bA[c];
    if (numB) v += numB[i] / (denB[(long)n * 4 + h] + 1e-16f) + bB[c];
    if (relu) v = fmaxf(v, 0.f);
    out[i] = v;
}

// ---------------- host launch ----------------
extern "C" void kernel_launch(void* const* d_in, const int* in_sizes, int n_in,
                              void* d_out, int out_size, void* d_ws, size_t ws_size,
                              hipStream_t stream) {
    const float* xp = (const float*)d_in[0];
    const float* xa = (const float*)d_in[1];
    const int* src_pp = (const int*)d_in[2];
    const int* dst_pp = (const int*)d_in[3];
    const int* src_ap = (const int*)d_in[4];
    const int* dst_ap = (const int*)d_in[5];
    const int* src_pa = (const int*)d_in[6];
    const int* dst_pa = (const int*)d_in[7];
    const float* Wsrc1 = (const float*)d_in[8];
    const float* Wdst1 = (const float*)d_in[9];
    const float* asrc1 = (const float*)d_in[10];
    const float* adst1 = (const float*)d_in[11];
    const float* b1    = (const float*)d_in[12];
    const float* Wsrc2 = (const float*)d_in[13];
    const float* Wdst2 = (const float*)d_in[14];
    const float* asrc2 = (const float*)d_in[15];
    const float* adst2 = (const float*)d_in[16];
    const float* b2    = (const float*)d_in[17];

    const int NP  = in_sizes[0] / 128;
    const int NA  = in_sizes[1] / 128;
    const int Epp = in_sizes[2];
    const int Eap = in_sizes[4];
    const int Epa = in_sizes[6];

    char* ws = (char*)d_ws;
    size_t off = 0;
    auto alloc = [&](size_t bytes) { size_t o = off; off = (off + bytes + 255) & ~(size_t)255; return o; };

    size_t o_wt1  = alloc(3L * 128 * 128 * 2);          // bf16 Wsrc1^T
    size_t o_wt2  = alloc(3L * 64 * 128 * 2);           // bf16 Wsrc2^T
    size_t o_wts1 = alloc(3L * 4 * 128 * 4);
    size_t o_wtd1 = alloc(3L * 4 * 128 * 4);
    size_t o_wts2 = alloc(3L * 4 * 128 * 4);
    size_t o_wtd2 = alloc(3L * 4 * 128 * 4);
    size_t o_xpb  = alloc((size_t)NP * 128 * 2);        // bf16 xp (layer2: bf16 h1p)
    size_t o_xab  = alloc((size_t)NA * 128 * 2);        // bf16 xa (layer2: bf16 h1a)
    size_t o_hsA  = alloc((size_t)NP * 128 * 4);        // hs pp  (layer2 reuses, CH=64)
    size_t o_hsB  = alloc((size_t)NA * 128 * 4);        // hs ap
    size_t o_hsC  = alloc((size_t)NP * 128 * 4);        // hs pa
    size_t o_esA  = alloc((size_t)NP * 16);
    size_t o_edA  = alloc((size_t)NP * 16);
    size_t o_esB  = alloc((size_t)NA * 16);
    size_t o_edB  = alloc((size_t)NP * 16);
    size_t o_esC  = alloc((size_t)NP * 16);
    size_t o_edC  = alloc((size_t)NA * 16);
    size_t zstart = off;                                // --- zeroed each layer ---
    size_t o_numA = alloc((size_t)NP * 128 * 4);
    size_t o_numB = alloc((size_t)NP * 128 * 4);
    size_t o_numC = alloc((size_t)NA * 128 * 4);
    size_t o_denA = alloc((size_t)NP * 16);
    size_t o_denB = alloc((size_t)NP * 16);
    size_t o_denC = alloc((size_t)NA * 16);
    size_t o_mA   = alloc((size_t)NP * 16);
    size_t o_mB   = alloc((size_t)NP * 16);
    size_t o_mC   = alloc((size_t)NA * 16);
    size_t zbytes = off - zstart;                       // --- end zero region ---
    size_t o_h1p  = alloc((size_t)NP * 128 * 4);
    size_t o_h1a  = alloc((size_t)NA * 128 * 4);

    unsigned short* wt1 = (unsigned short*)(ws + o_wt1);
    unsigned short* wt2 = (unsigned short*)(ws + o_wt2);
    float* wts1 = (float*)(ws + o_wts1); float* wtd1 = (float*)(ws + o_wtd1);
    float* wts2 = (float*)(ws + o_wts2); float* wtd2 = (float*)(ws + o_wtd2);
    unsigned short* xpb = (unsigned short*)(ws + o_xpb);
    unsigned short* xab = (unsigned short*)(ws + o_xab);
    float* hsA = (float*)(ws + o_hsA); float* hsB = (float*)(ws + o_hsB); float* hsC = (float*)(ws + o_hsC);
    float* esA = (float*)(ws + o_esA); float* edA = (float*)(ws + o_edA);
    float* esB = (float*)(ws + o_esB); float* edB = (float*)(ws + o_edB);
    float* esC = (float*)(ws + o_esC); float* edC = (float*)(ws + o_edC);
    float* numA = (float*)(ws + o_numA); float* numB = (float*)(ws + o_numB); float* numC = (float*)(ws + o_numC);
    float* denA = (float*)(ws + o_denA); float* denB = (float*)(ws + o_denB); float* denC = (float*)(ws + o_denC);
    unsigned* mA = (unsigned*)(ws + o_mA); unsigned* mB = (unsigned*)(ws + o_mB); unsigned* mC = (unsigned*)(ws + o_mC);
    float* h1p = (float*)(ws + o_h1p); float* h1a = (float*)(ws + o_h1a);
    float* out = (float*)d_out;

    auto cdiv = [](long a, long b) { return (int)((a + b - 1) / b); };

    // ---- weight prep ----
    k_wt_bf16<<<cdiv(3L * 128 * 128, 256), 256, 0, stream>>>(Wsrc1, wt1, 128, 128);
    k_wt_bf16<<<cdiv(3L * 128 * 64, 256), 256, 0, stream>>>(Wsrc2, wt2, 128, 64);
    k_wtilde<<<cdiv(3 * 4 * 128, 256), 256, 0, stream>>>(Wsrc1, asrc1, wts1, 128, 128, 32);
    k_wtilde<<<cdiv(3 * 4 * 128, 256), 256, 0, stream>>>(Wdst1, adst1, wtd1, 128, 128, 32);
    k_wtilde<<<cdiv(3 * 4 * 128, 256), 256, 0, stream>>>(Wsrc2, asrc2, wts2, 128, 64, 16);
    k_wtilde<<<cdiv(3 * 4 * 128, 256), 256, 0, stream>>>(Wdst2, adst2, wtd2, 128, 64, 16);

    // ================= layer 1 (CH=128) =================
    hipMemsetAsync(ws + zstart, 0, zbytes, stream);
    k_f2bf<<<cdiv((long)NP * 128 / 4, 256), 256, 0, stream>>>(xp, xpb, (long)NP * 128);
    k_f2bf<<<cdiv((long)NA * 128 / 4, 256), 256, 0, stream>>>(xa, xab, (long)NA * 128);
    k_gemm<128><<<cdiv(NP, 128), 256, 0, stream>>>(xpb, wt1,                  hsA, NP);
    k_gemm<128><<<cdiv(NA, 128), 256, 0, stream>>>(xab, wt1 + 1L * 128 * 128, hsB, NA);
    k_gemm<128><<<cdiv(NP, 128), 256, 0, stream>>>(xpb, wt1 + 2L * 128 * 128, hsC, NP);
    k_escalar<<<cdiv((long)NP * 4, 256), 256, 0, stream>>>(xp, wts1,           esA, NP);
    k_escalar<<<cdiv((long)NP * 4, 256), 256, 0, stream>>>(xp, wtd1,           edA, NP);
    k_escalar<<<cdiv((long)NA * 4, 256), 256, 0, stream>>>(xa, wts1 + 4 * 128, esB, NA);
    k_escalar<<<cdiv((long)NP * 4, 256), 256, 0, stream>>>(xp, wtd1 + 4 * 128, edB, NP);
    k_escalar<<<cdiv((long)NP * 4, 256), 256, 0, stream>>>(xp, wts1 + 8 * 128, esC, NP);
    k_escalar<<<cdiv((long)NA * 4, 256), 256, 0, stream>>>(xa, wtd1 + 8 * 128, edC, NA);
    k_edge_max<<<cdiv(Epp, 256), 256, 0, stream>>>(src_pp, dst_pp, esA, edA, mA, Epp);
    k_edge_max<<<cdiv(Eap, 256), 256, 0, stream>>>(src_ap, dst_ap, esB, edB, mB, Eap);
    k_edge_max<<<cdiv(Epa, 256), 256, 0, stream>>>(src_pa, dst_pa, esC, edC, mC, Epa);
    k_edge_scatter<128><<<cdiv(Epp, 8), 256, 0, stream>>>(src_pp, dst_pp, esA, edA, mA, hsA, numA, denA, Epp);
    k_edge_scatter<128><<<cdiv(Eap, 8), 256, 0, stream>>>(src_ap, dst_ap, esB, edB, mB, hsB, numB, denB, Eap);
    k_edge_scatter<128><<<cdiv(Epa, 8), 256, 0, stream>>>(src_pa, dst_pa, esC, edC, mC, hsC, numC, denC, Epa);
    k_finalize<<<cdiv((long)NP * 128, 256), 256, 0, stream>>>(numA, denA, b1, numB, denB, b1 + 128, h1p, NP, 128, 1);
    k_finalize<<<cdiv((long)NA * 128, 256), 256, 0, stream>>>(numC, denC, b1 + 256, nullptr, nullptr, nullptr, h1a, NA, 128, 1);

    // ================= layer 2 (CH=64) =================
    hipMemsetAsync(ws + zstart, 0, zbytes, stream);
    k_f2bf<<<cdiv((long)NP * 128 / 4, 256), 256, 0, stream>>>(h1p, xpb, (long)NP * 128);   // reuse bf16 buffers
    k_f2bf<<<cdiv((long)NA * 128 / 4, 256), 256, 0, stream>>>(h1a, xab, (long)NA * 128);
    k_gemm<64><<<cdiv(NP, 128), 256, 0, stream>>>(xpb, wt2,                 hsA, NP);
    k_gemm<64><<<cdiv(NA, 128), 256, 0, stream>>>(xab, wt2 + 1L * 64 * 128, hsB, NA);
    k_gemm<64><<<cdiv(NP, 128), 256, 0, stream>>>(xpb, wt2 + 2L * 64 * 128, hsC, NP);
    k_escalar<<<cdiv((long)NP * 4, 256), 256, 0, stream>>>(h1p, wts2,           esA, NP);
    k_escalar<<<cdiv((long)NP * 4, 256), 256, 0, stream>>>(h1p, wtd2,           edA, NP);
    k_escalar<<<cdiv((long)NA * 4, 256), 256, 0, stream>>>(h1a, wts2 + 4 * 128, esB, NA);
    k_escalar<<<cdiv((long)NP * 4, 256), 256, 0, stream>>>(h1p, wtd2 + 4 * 128, edB, NP);
    k_escalar<<<cdiv((long)NP * 4, 256), 256, 0, stream>>>(h1p, wts2 + 8 * 128, esC, NP);
    k_escalar<<<cdiv((long)NA * 4, 256), 256, 0, stream>>>(h1a, wtd2 + 8 * 128, edC, NA);
    k_edge_max<<<cdiv(Epp, 256), 256, 0, stream>>>(src_pp, dst_pp, esA, edA, mA, Epp);
    k_edge_max<<<cdiv(Eap, 256), 256, 0, stream>>>(src_ap, dst_ap, esB, edB, mB, Eap);
    k_edge_max<<<cdiv(Epa, 256), 256, 0, stream>>>(src_pa, dst_pa, esC, edC, mC, Epa);
    k_edge_scatter<64><<<cdiv(Epp, 8), 256, 0, stream>>>(src_pp, dst_pp, esA, edA, mA, hsA, numA, denA, Epp);
    k_edge_scatter<64><<<cdiv(Eap, 8), 256, 0, stream>>>(src_ap, dst_ap, esB, edB, mB, hsB, numB, denB, Eap);
    k_edge_scatter<64><<<cdiv(Epa, 8), 256, 0, stream>>>(src_pa, dst_pa, esC, edC, mC, hsC, numC, denC, Epa);
    k_finalize<<<cdiv((long)NP * 64, 256), 256, 0, stream>>>(numA, denA, b2, numB, denB, b2 + 64, out, NP, 64, 0);
    k_finalize<<<cdiv((long)NA * 64, 256), 256, 0, stream>>>(numC, denC, b2 + 128, nullptr, nullptr, nullptr,
                                                             out + (size_t)NP * 64, NA, 64, 0);
}
